// CommonFeatureExtractor_69887707840829
// MI455X (gfx1250) — matrix operand order repaired and checked by reading, not compiled, not run on hardware
//
#include <hip/hip_runtime.h>

typedef __attribute__((ext_vector_type(16))) __bf16 v16bf;
typedef __attribute__((ext_vector_type(8)))  __bf16 v8bf;
typedef __attribute__((ext_vector_type(8)))  float  v8f;

#define BM 128
#define BN 128
#define BK 32
#define BKP 40   // padded LDS row stride (ushorts): 80 B row, 16B-aligned fragments

__device__ __forceinline__ unsigned short f2bf(float f) {
  unsigned u = __float_as_uint(f);
  return (unsigned short)((u + 0x7FFFu + ((u >> 16) & 1u)) >> 16);  // RNE
}
__device__ __forceinline__ float bf2f(unsigned short h) {
  return __uint_as_float((unsigned)h << 16);
}

// Async global->LDS copy, 16 bytes per lane. VDST = LDS byte offset (low 32 bits
// of the flat LDS address), VADDR = 64-bit global address. Tracked by ASYNCcnt.
__device__ __forceinline__ void async_b128(unsigned lds, const void* g) {
  asm volatile("global_load_async_to_lds_b128 %0, %1, off"
               :: "v"(lds), "v"((unsigned long long)g) : "memory");
}

union V16U { v16bf v; v8bf h[2]; };

// C[4096, N] = act(A[4096, K] @ Wt[N, K]^T + bias); A, Wt bf16 (K % 32 == 0).
// ACT: 0=none, 1=relu, 2=gate*sigmoid. OUT32: 1 -> f32 out, else bf16 out.
template<int ACT, int OUT32>
__global__ __launch_bounds__(256)
void gemm_bf16(const unsigned short* __restrict__ A,
               const unsigned short* __restrict__ Wt,
               const float* __restrict__ bias,
               const unsigned short* __restrict__ gate,
               unsigned short* __restrict__ C16, float* __restrict__ C32,
               int ldc, int K, int N)
{
  __shared__ __align__(16) unsigned short sA[2][BM * BKP];
  __shared__ __align__(16) unsigned short sW[2][BN * BKP];

  const int tid  = threadIdx.x;
  const int bm   = blockIdx.x * BM;
  const int bn   = blockIdx.y * BN;
  const int lane = tid & 31;
  const int wave = tid >> 5;
  const int rbase = (wave >> 1) * 32;   // 4 waves along M
  const int cbase = (wave & 1) * 64;    // 2 waves along N
  const int lr   = lane & 15;
  const int hiL  = lane >> 4;
  const int kha  = hiL * 8;             // A-frag K base (16-bit A layout)
  const int khb  = hiL * 8;             // B-frag K base (same layout as A)

  // Staging: 512 chunks of 16 B per tile (128 rows x 4 chunks); 2 chunks/thread
  // for A and 2 for Wt. row = chunk>>2, q = chunk&3 (q*8 elements into the row).
  const int r0 = tid >> 2,  q0 = tid & 3;
  const int r1 = r0 + 64;   // chunk tid+256: same q
  const unsigned short* gA0 = A  + (size_t)(bm + r0) * K + q0 * 8;
  const unsigned short* gA1 = A  + (size_t)(bm + r1) * K + q0 * 8;
  const unsigned short* gW0 = Wt + (size_t)(bn + r0) * K + q0 * 8;
  const unsigned short* gW1 = Wt + (size_t)(bn + r1) * K + q0 * 8;

  auto stage = [&](int buf, int k0) {
    async_b128((unsigned)(size_t)&sA[buf][r0 * BKP + q0 * 8], gA0 + k0);
    async_b128((unsigned)(size_t)&sA[buf][r1 * BKP + q0 * 8], gA1 + k0);
    async_b128((unsigned)(size_t)&sW[buf][r0 * BKP + q0 * 8], gW0 + k0);
    async_b128((unsigned)(size_t)&sW[buf][r1 * BKP + q0 * 8], gW1 + k0);
  };

  v8f acc[2][4];
  {
    v8f z = {};
    #pragma unroll
    for (int i = 0; i < 2; ++i)
      #pragma unroll
      for (int j = 0; j < 4; ++j) acc[i][j] = z;
  }

  const int nk = K / BK;
  stage(0, 0);
  for (int kt = 0; kt < nk; ++kt) {
    const int cur = kt & 1;
    if (kt + 1 < nk) {
      __syncthreads();                       // all waves done reading buf cur^1
      stage(cur ^ 1, (kt + 1) * BK);         // prefetch next tile (async)
      asm volatile("s_wait_asynccnt 0x4" ::: "memory");  // this tile's 4 done
    } else {
      asm volatile("s_wait_asynccnt 0x0" ::: "memory");
    }
    __syncthreads();                         // tile kt visible from all waves

    v16bf afrag[2], bfrag[4];
    #pragma unroll
    for (int mt = 0; mt < 2; ++mt) {
      const v8bf* p = (const v8bf*)&sA[cur][(size_t)(rbase + mt * 16 + lr) * BKP + kha];
      V16U u; u.h[0] = p[0]; u.h[1] = p[2];  // K = kha..+7, kha+16..+23
      afrag[mt] = u.v;
    }
    #pragma unroll
    for (int nt = 0; nt < 4; ++nt) {
      const v8bf* p = (const v8bf*)&sW[cur][(size_t)(cbase + nt * 16 + lr) * BKP + khb];
      V16U u; u.h[0] = p[0]; u.h[1] = p[2];  // K = khb..+7, khb+16..+23
      bfrag[nt] = u.v;
    }
    #pragma unroll
    for (int mt = 0; mt < 2; ++mt)
      #pragma unroll
      for (int nt = 0; nt < 4; ++nt)
        acc[mt][nt] = __builtin_amdgcn_wmma_f32_16x16x32_bf16(
            false, afrag[mt], false, bfrag[nt], (short)0, acc[mt][nt], false, false);
  }

  #pragma unroll
  for (int mt = 0; mt < 2; ++mt) {
    int rtop = bm + rbase + mt * 16 + hiL * 8;
    #pragma unroll
    for (int nt = 0; nt < 4; ++nt) {
      int col = bn + cbase + nt * 16 + lr;
      float bv = bias[col];
      #pragma unroll
      for (int v = 0; v < 8; ++v) {
        int row = rtop + v;
        float x = acc[mt][nt][v] + bv;
        if (ACT == 1) x = fmaxf(x, 0.f);
        if (ACT == 2) {
          float g = bf2f(gate[(size_t)row * N + col]);
          x = g / (1.f + __expf(-x));
        }
        if (OUT32) C32[(size_t)row * ldc + col] = x;
        else       C16[(size_t)row * ldc + col] = f2bf(x);
      }
    }
  }
}

// De-interleave one fp_features slice to bf16, zero-padded to Kpad columns.
__global__ __launch_bounds__(256)
void slice_to_bf16(const float* __restrict__ X, int soff, int Kreal, int Kpad,
                   unsigned short* __restrict__ dst, int total)
{
  int idx = blockIdx.x * 256 + threadIdx.x;
  if (idx >= total) return;
  int b = idx / Kpad, k = idx - b * Kpad;
  float v = (k < Kreal) ? X[(size_t)b * 6338 + soff + k] : 0.f;
  dst[idx] = f2bf(v);
}

// Transpose-convert W[Kreal, N] f32 -> Wt[N, Kpad] bf16 (zero-padded K).
__global__ __launch_bounds__(256)
void weight_to_bf16_t(const float* __restrict__ W, int Kreal, int N, int Kpad,
                      unsigned short* __restrict__ Wt, int total)
{
  int idx = blockIdx.x * 256 + threadIdx.x;
  if (idx >= total) return;
  int n = idx / Kpad, k = idx - n * Kpad;
  float v = (k < Kreal) ? W[(size_t)k * N + n] : 0.f;
  Wt[idx] = f2bf(v);
}

// Per-row fusion (bf16 in/out): norms, 10 pairwise cosines, pair softmax,
// common feature, wg logits + softmax, weighted fps sum. One block per row.
__global__ __launch_bounds__(256)
void fuse_rows(const unsigned short* __restrict__ fps,   // [B,5,512] bf16
               const float* __restrict__ wg_w,           // [2560,5]
               const float* __restrict__ wg_b,           // [5]
               unsigned short* __restrict__ common,      // [B,512] bf16
               unsigned short* __restrict__ cat)         // [B,1024] bf16
{
  const int b   = blockIdx.x;
  const int tid = threadIdx.x;
  const unsigned short* f = fps + (size_t)b * 2560;

  float p[20];
  #pragma unroll
  for (int q = 0; q < 20; ++q) p[q] = 0.f;

  for (int h = tid; h < 512; h += 256) {
    float x[5];
    #pragma unroll
    for (int i = 0; i < 5; ++i) x[i] = bf2f(f[i * 512 + h]);
    #pragma unroll
    for (int i = 0; i < 5; ++i) p[i] += x[i] * x[i];
    int q = 5;
    #pragma unroll
    for (int i = 0; i < 5; ++i)
      #pragma unroll
      for (int j = i + 1; j < 5; ++j) p[q++] += x[i] * x[j];
  }
  for (int t = tid; t < 2560; t += 256) {
    float xv = bf2f(f[t]);
    #pragma unroll
    for (int j = 0; j < 5; ++j) p[15 + j] += xv * wg_w[t * 5 + j];
  }

  __shared__ float red[8 * 20];
  __shared__ float fin[20];
  #pragma unroll
  for (int q = 0; q < 20; ++q) {
    float v = p[q];
    for (int off = 16; off > 0; off >>= 1) v += __shfl_xor(v, off, 32);
    if ((tid & 31) == 0) red[(tid >> 5) * 20 + q] = v;
  }
  __syncthreads();
  if (tid < 20) {
    float s = 0.f;
    for (int w = 0; w < 8; ++w) s += red[w * 20 + tid];
    fin[tid] = s;
  }
  __syncthreads();

  float nrm[5];
  #pragma unroll
  for (int i = 0; i < 5; ++i) nrm[i] = fmaxf(sqrtf(fmaxf(fin[i], 0.f)), 1e-12f);

  float wp[10]; float S = 0.f;
  {
    int q = 5;
    #pragma unroll
    for (int i = 0; i < 5; ++i)
      #pragma unroll
      for (int j = i + 1; j < 5; ++j) {
        float sim = fin[q] / (nrm[i] * nrm[j]);
        float e = (sim > 0.f) ? __expf(sim) : 0.f;
        wp[q - 5] = e; S += e; ++q;
      }
  }
  const bool anySel = (S > 0.f);
  const float invS = 1.f / fmaxf(S, 1e-30f);
  #pragma unroll
  for (int q = 0; q < 10; ++q) wp[q] *= invS;

  float lg[5], mx = -1e30f;
  #pragma unroll
  for (int j = 0; j < 5; ++j) { lg[j] = fin[15 + j] + wg_b[j]; mx = fmaxf(mx, lg[j]); }
  float fw[5], fs = 0.f;
  #pragma unroll
  for (int j = 0; j < 5; ++j) { fw[j] = __expf(lg[j] - mx); fs += fw[j]; }
  const float invfs = 1.f / fs;
  #pragma unroll
  for (int j = 0; j < 5; ++j) fw[j] *= invfs;

  for (int h = tid; h < 512; h += 256) {
    float x[5];
    #pragma unroll
    for (int i = 0; i < 5; ++i) x[i] = bf2f(f[i * 512 + h]);
    float agg = 0.f, mean = 0.f, ws = 0.f;
    #pragma unroll
    for (int i = 0; i < 5; ++i) { mean += x[i]; ws += fw[i] * x[i]; }
    mean *= 0.2f;
    int q = 0;
    #pragma unroll
    for (int i = 0; i < 5; ++i)
      #pragma unroll
      for (int j = i + 1; j < 5; ++j) {
        float prod = x[i] * x[j];
        float cf = (prod > 0.f) ? 0.5f * (x[i] + x[j]) : 0.f;
        agg += wp[q++] * cf;
      }
    common[(size_t)b * 512 + h] = f2bf(anySel ? agg : mean);
    cat[(size_t)b * 1024 + h] = f2bf(ws);
  }
}

extern "C" void kernel_launch(void* const* d_in, const int* in_sizes, int n_in,
                              void* d_out, int out_size, void* d_ws, size_t ws_size,
                              hipStream_t stream) {
  (void)in_sizes; (void)n_in; (void)out_size; (void)ws_size;
  const float* X     = (const float*)d_in[0];
  const float* ap_w1 = (const float*)d_in[1];  const float* ap_b1 = (const float*)d_in[2];
  const float* ap_w2 = (const float*)d_in[3];  const float* ap_b2 = (const float*)d_in[4];
  const float* ma_w1 = (const float*)d_in[5];  const float* ma_b1 = (const float*)d_in[6];
  const float* ma_w2 = (const float*)d_in[7];  const float* ma_b2 = (const float*)d_in[8];
  const float* mb_w1 = (const float*)d_in[9];  const float* mb_b1 = (const float*)d_in[10];
  const float* mb_w2 = (const float*)d_in[11]; const float* mb_b2 = (const float*)d_in[12];
  const float* mc_w1 = (const float*)d_in[13]; const float* mc_b1 = (const float*)d_in[14];
  const float* mc_w2 = (const float*)d_in[15]; const float* mc_b2 = (const float*)d_in[16];
  const float* ph_w1 = (const float*)d_in[17]; const float* ph_b1 = (const float*)d_in[18];
  const float* ph_w2 = (const float*)d_in[19]; const float* ph_b2 = (const float*)d_in[20];
  const float* wg_w  = (const float*)d_in[21]; const float* wg_b  = (const float*)d_in[22];
  const float* enh_w = (const float*)d_in[23]; const float* enh_b = (const float*)d_in[24];
  const float* fus_w = (const float*)d_in[25]; const float* fus_b = (const float*)d_in[26];

  unsigned short* wsp = (unsigned short*)d_ws;
  auto alloc = [&](size_t elems) {
    unsigned short* p = wsp;
    wsp += (elems + 127) & ~(size_t)127;
    return p;
  };
  // bf16 activation slices (K padded to multiple of 32)
  unsigned short* Xap = alloc((size_t)4096 * 2048);
  unsigned short* Xma = alloc((size_t)4096 * 192);
  unsigned short* Xmb = alloc((size_t)4096 * 2048);
  unsigned short* Xmc = alloc((size_t)4096 * 2048);
  unsigned short* Xph = alloc((size_t)4096 * 32);
  // bf16 transposed weights [N, Kpad]
  unsigned short* apw1t = alloc((size_t)512 * 2048);
  unsigned short* apw2t = alloc((size_t)512 * 512);
  unsigned short* maw1t = alloc((size_t)256 * 192);
  unsigned short* maw2t = alloc((size_t)512 * 256);
  unsigned short* mbw1t = alloc((size_t)512 * 2048);
  unsigned short* mbw2t = alloc((size_t)512 * 512);
  unsigned short* mcw1t = alloc((size_t)512 * 2048);
  unsigned short* mcw2t = alloc((size_t)512 * 512);
  unsigned short* phw1t = alloc((size_t)128 * 32);
  unsigned short* phw2t = alloc((size_t)512 * 128);
  unsigned short* enhwt = alloc((size_t)512 * 512);
  unsigned short* fuswt = alloc((size_t)512 * 1024);
  // bf16 intermediates
  unsigned short* h      = alloc((size_t)4096 * 512);
  unsigned short* fps    = alloc((size_t)4096 * 2560);
  unsigned short* common = alloc((size_t)4096 * 512);
  unsigned short* cat    = alloc((size_t)4096 * 1024);
  float* out = (float*)d_out;

  auto cvt_slice = [&](int soff, int Kr, int Kp, unsigned short* dst) {
    int total = 4096 * Kp;
    slice_to_bf16<<<dim3((total + 255) / 256), dim3(256), 0, stream>>>(X, soff, Kr, Kp, dst, total);
  };
  auto cvt_w = [&](const float* Wf, int Kr, int N, int Kp, unsigned short* dst) {
    int total = N * Kp;
    weight_to_bf16_t<<<dim3((total + 255) / 256), dim3(256), 0, stream>>>(Wf, Kr, N, Kp, dst, total);
  };
  auto gemm = [&](int act, int out32, const unsigned short* A, const unsigned short* Wt,
                  const float* bias, const unsigned short* gate,
                  unsigned short* C16, float* C32, int ldc, int K, int N) {
    dim3 grid(4096 / BM, N / BN), blk(256);
    if (out32)
      gemm_bf16<0, 1><<<grid, blk, 0, stream>>>(A, Wt, bias, gate, C16, C32, ldc, K, N);
    else if (act == 1)
      gemm_bf16<1, 0><<<grid, blk, 0, stream>>>(A, Wt, bias, gate, C16, C32, ldc, K, N);
    else if (act == 2)
      gemm_bf16<2, 0><<<grid, blk, 0, stream>>>(A, Wt, bias, gate, C16, C32, ldc, K, N);
    else
      gemm_bf16<0, 0><<<grid, blk, 0, stream>>>(A, Wt, bias, gate, C16, C32, ldc, K, N);
  };

  // one-time bf16 conversion / padding / weight transposition
  cvt_slice(0,    2048, 2048, Xap);
  cvt_slice(2048, 167,  192,  Xma);
  cvt_slice(2215, 2048, 2048, Xmb);
  cvt_slice(4263, 2048, 2048, Xmc);
  cvt_slice(6311, 27,   32,   Xph);
  cvt_w(ap_w1, 2048, 512, 2048, apw1t);  cvt_w(ap_w2, 512,  512, 512,  apw2t);
  cvt_w(ma_w1, 167,  256, 192,  maw1t);  cvt_w(ma_w2, 256,  512, 256,  maw2t);
  cvt_w(mb_w1, 2048, 512, 2048, mbw1t);  cvt_w(mb_w2, 512,  512, 512,  mbw2t);
  cvt_w(mc_w1, 2048, 512, 2048, mcw1t);  cvt_w(mc_w2, 512,  512, 512,  mcw2t);
  cvt_w(ph_w1, 27,   128, 32,   phw1t);  cvt_w(ph_w2, 128,  512, 128,  phw2t);
  cvt_w(enh_w, 512,  512, 512,  enhwt);  cvt_w(fus_w, 1024, 512, 1024, fuswt);

  // 5 encoders: Linear->ReLU->Linear, fps strided into [B,5,512]
  gemm(1, 0, Xap, apw1t, ap_b1, nullptr, h,          nullptr, 512,  2048, 512);
  gemm(0, 0, h,   apw2t, ap_b2, nullptr, fps + 0,    nullptr, 2560, 512,  512);
  gemm(1, 0, Xma, maw1t, ma_b1, nullptr, h,          nullptr, 256,  192,  256);
  gemm(0, 0, h,   maw2t, ma_b2, nullptr, fps + 512,  nullptr, 2560, 256,  512);
  gemm(1, 0, Xmb, mbw1t, mb_b1, nullptr, h,          nullptr, 512,  2048, 512);
  gemm(0, 0, h,   mbw2t, mb_b2, nullptr, fps + 1024, nullptr, 2560, 512,  512);
  gemm(1, 0, Xmc, mcw1t, mc_b1, nullptr, h,          nullptr, 512,  2048, 512);
  gemm(0, 0, h,   mcw2t, mc_b2, nullptr, fps + 1536, nullptr, 2560, 512,  512);
  gemm(1, 0, Xph, phw1t, ph_b1, nullptr, h,          nullptr, 128,  32,   128);
  gemm(0, 0, h,   phw2t, ph_b2, nullptr, fps + 2048, nullptr, 2560, 128,  512);

  // pairwise-cosine fusion + wg softmax -> common, cat[:, :512]
  fuse_rows<<<dim3(4096), dim3(256), 0, stream>>>(fps, wg_w, wg_b, common, cat);

  // enhanced = common * sigmoid(common @ enh_w + enh_b) -> cat[:, 512:]
  gemm(2, 0, common, enhwt, enh_b, common, cat + 512, nullptr, 1024, 512, 512);
  // fused = cat @ fus_w + fus_b -> d_out (f32)
  gemm(0, 1, cat, fuswt, fus_b, nullptr, nullptr, out, 512, 1024, 512);
}
